// LSTM_DOUBLE_ATT_57990648430687
// MI455X (gfx1250) — matrix operand order repaired and checked by reading, not compiled, not run on hardware
//
#include <hip/hip_runtime.h>
#include <hip/hip_bf16.h>
#include <math.h>

typedef __attribute__((ext_vector_type(16))) _Float16 v16h;
typedef __attribute__((ext_vector_type(8)))  float    v8f;
typedef unsigned int u32x4 __attribute__((ext_vector_type(4)));
typedef int          i32x4 __attribute__((ext_vector_type(4)));
typedef int          i32x8 __attribute__((ext_vector_type(8)));

#define B_  256
#define I_  512
#define R_  512
#define A_  196
#define V_  10000
#define H4_ 2048
#define LDP 36   // LDS row stride in halfs: 32 data + 4 pad (matches TDM pad 16DW+2DW)

// ---------------- f32x2 -> packed f16x2 ----------------
static __device__ __forceinline__ unsigned int pack_f16x2(float a, float b) {
  union { _Float16 h[2]; unsigned int u; } u;
  u.h[0] = (_Float16)a; u.h[1] = (_Float16)b;
  return u.u;
}

// ---------------- TDM: DMA a tile_rows x 32 f16 tile (row stride ld_elems) to LDS ----
// D# per CDNA5 ISA. data_size=2B; pad_interval=16 DWORDs (one
// 32-half row), pad_amount=2 DWORDs (4 halfs) -> LDP=36 rows. Rows beyond rows_valid
// hardware zero-filled (tensor_dim1 < tile_dim1). Tracked by TENSORcnt, in-order.
static __device__ __forceinline__ void tdm_load_tile_f16(
    unsigned lds_off, const _Float16* gsrc, unsigned ld_elems,
    unsigned rows_valid, unsigned tile_rows)
{
  unsigned long long ga = (unsigned long long)(size_t)gsrc;
  u32x4 g0;
  g0.x = 1u;                                                   // count=1 (valid user D#)
  g0.y = lds_off;                                              // lds_addr (bytes)
  g0.z = (unsigned)ga;                                         // global_addr[31:0]
  g0.w = (unsigned)((ga >> 32) & 0x01ffffffull) | 0x80000000u; // addr[56:32] | type=2
  i32x8 g1;
  const unsigned td0 = 32u;                                    // tensor_dim0 (cols)
  g1[0] = (int)((1u << 16) | (1u << 20) | (3u << 22) | (1u << 25)); // dsz=2B,pad_en,16DW,+2DW
  g1[1] = (int)((td0 & 0xffffu) << 16);                        // tensor_dim0[15:0]
  g1[2] = (int)(((td0 >> 16) & 0xffffu) | ((rows_valid & 0xffffu) << 16));
  g1[3] = (int)(((rows_valid >> 16) & 0xffffu) | (32u << 16)); // tile_dim0=32
  g1[4] = (int)(tile_rows & 0xffffu);                          // tile_dim1
  g1[5] = (int)ld_elems;                                       // tensor_dim0_stride lo
  g1[6] = 0;
  g1[7] = 0;
  i32x4 z4 = {};
#if defined(__clang_major__) && (__clang_major__ >= 23)
  i32x8 z8 = {};
  __builtin_amdgcn_tensor_load_to_lds(g0, g1, z4, z4, z8, 0);
#else
  __builtin_amdgcn_tensor_load_to_lds(g0, g1, z4, z4, 0);
#endif
}

// ---------------- branch-free VALU stager for f32 activations (rows always valid) ----
template<int ROWS>
static __device__ __forceinline__ void load_tile(
    const float* __restrict__ src, int ld, int row0, int kk,
    _Float16* __restrict__ dst, int tid)
{
  constexpr int ITER = (ROWS * 8) / 256;
  float4 tmp[ITER];
#pragma unroll
  for (int i = 0; i < ITER; ++i) {
    int idx = tid + i * 256;
    tmp[i] = *(const float4*)(src + (size_t)(row0 + (idx >> 3)) * ld + kk +
                              ((idx & 7) << 2));
  }
#pragma unroll
  for (int i = 0; i < ITER; ++i) {
    int idx = tid + i * 256;
    *(uint2*)(dst + (idx >> 3) * LDP + ((idx & 7) << 2)) =
        make_uint2(pack_f16x2(tmp[i].x, tmp[i].y), pack_f16x2(tmp[i].z, tmp[i].w));
  }
}

// ---------------- WMMA fragment helpers (wave32, 16x16x32 f16) ----------------
static __device__ __forceinline__ v16h frag_a(const _Float16* tile, int lane) {
  const int m  = lane & 15;
  const int kb = (lane < 16) ? 0 : 8;
  const _Float16* row = tile + m * LDP;
  v16h a;
#pragma unroll
  for (int e = 0; e < 8; ++e) a[e] = row[kb + e];
#pragma unroll
  for (int e = 0; e < 8; ++e) a[8 + e] = row[kb + 16 + e];
  return a;
}
static __device__ __forceinline__ v16h frag_b(const _Float16* tile, int lane) {
  const int n  = lane & 15;
  const int kb = (lane < 16) ? 0 : 8;
  const _Float16* row = tile + n * LDP;
  v16h b;
#pragma unroll
  for (int e = 0; e < 16; ++e) b[e] = row[kb + (e < 8 ? e : e + 8)];
  return b;
}
static __device__ __forceinline__ v8f wmma_f16(v16h a, v16h b, v8f c) {
  return __builtin_amdgcn_wmma_f32_16x16x32_f16(false, a, false, b, (short)0, c,
                                                false, false);
}
// D: lanes 0-15: N=lane, M=j; lanes 16-31: N=lane-16, M=8+j.

// ---------------- block reduction (256 threads = 8 waves) ----------------
static __device__ __forceinline__ float blk_reduce(float v, bool is_max, float* red) {
#pragma unroll
  for (int s = 16; s > 0; s >>= 1) {
    float o = __shfl_xor(v, s, 32);
    v = is_max ? fmaxf(v, o) : (v + o);
  }
  const int lane = threadIdx.x & 31, wv = threadIdx.x >> 5;
  if (lane == 0) red[wv] = v;
  __syncthreads();
  if (threadIdx.x == 0) {
    float r = red[0];
#pragma unroll
    for (int i = 1; i < 8; ++i) r = is_max ? fmaxf(r, red[i]) : (r + red[i]);
    red[0] = r;
  }
  __syncthreads();
  float r = red[0];
  __syncthreads();
  return r;
}

// ---------------- f32 -> f16 conversion (one-time, per matrix) ----------------
__global__ __launch_bounds__(256) void k_f32_to_f16(
    const float* __restrict__ src, _Float16* __restrict__ dst, int n)
{
  int i = (blockIdx.x * 256 + threadIdx.x) * 8;
  if (i + 8 <= n) {
    float4 a = *(const float4*)(src + i);
    float4 b = *(const float4*)(src + i + 4);
    uint4 p;
    p.x = pack_f16x2(a.x, a.y); p.y = pack_f16x2(a.z, a.w);
    p.z = pack_f16x2(b.x, b.y); p.w = pack_f16x2(b.z, b.w);
    *(uint4*)(dst + i) = p;
  }
}

// ---------------- WMMA GEMM: Out = X(MxK,f32) @ W16(NxK,f16)^T + bias ----------------
// 128(M) x 64(N) block tile; W tile TDM double-buffered, X tile VALU-converted.
// M is always a multiple of 128 here (M=256) -> no row guards.
__global__ __launch_bounds__(256) void k_gemm(
    const float* __restrict__ X, const _Float16* __restrict__ W16,
    const float* __restrict__ bias, float* __restrict__ Out,
    int M, int N, int K)
{
  __shared__ __align__(16) _Float16 Xs[128 * LDP];
  __shared__ __align__(16) _Float16 Ws[2][64 * LDP];
  const int tid  = threadIdx.x;
  const int lane = tid & 31;
  const int wv   = tid >> 5;
  const int mb   = blockIdx.y * 128;
  const int nb   = blockIdx.x * 64;
  const unsigned ws_off0 = (unsigned)(size_t)(&Ws[0][0]);
  const unsigned ws_off1 = (unsigned)(size_t)(&Ws[1][0]);
  v8f acc[4];
  { v8f z = {};
#pragma unroll
    for (int t = 0; t < 4; ++t) acc[t] = z; }

  if (wv == 0)   // prologue: DMA first W slab into buffer 0
    tdm_load_tile_f16(ws_off0, W16 + (size_t)nb * K, (unsigned)K,
                      (unsigned)(N - nb), 64u);
  int buf = 0;
  for (int kk = 0; kk < K; kk += 32) {
    if (wv == 0) {
      if (kk + 32 < K) {   // DMA next slab into alternate buffer, overlap with WMMA
        tdm_load_tile_f16(buf ? ws_off0 : ws_off1,
                          W16 + (size_t)nb * K + kk + 32, (unsigned)K,
                          (unsigned)(N - nb), 64u);
        __builtin_amdgcn_s_wait_tensorcnt(1);   // current slab landed
      } else {
        __builtin_amdgcn_s_wait_tensorcnt(0);
      }
    }
    load_tile<128>(X, K, mb, kk, Xs, tid);
    __syncthreads();
    v16h af = frag_a(&Xs[wv * 16 * LDP], lane);
    const _Float16* wsb = &Ws[buf][0];
#pragma unroll
    for (int nt = 0; nt < 4; ++nt) {
      v16h bf = frag_b(wsb + nt * 16 * LDP, lane);
      acc[nt] = wmma_f16(af, bf, acc[nt]);
    }
    __syncthreads();
    buf ^= 1;
  }

  const int nloc = lane & 15;
  const int moff = (lane < 16) ? 0 : 8;
#pragma unroll
  for (int nt = 0; nt < 4; ++nt) {
    int gn = nb + nt * 16 + nloc;
    if (gn < N) {
      float bv = bias[gn];
#pragma unroll
      for (int j = 0; j < 8; ++j) {
        int gm = mb + wv * 16 + j + moff;
        Out[(size_t)gm * N + gn] = acc[nt][j] + bv;
      }
    }
  }
}

// ---------------- fused 3-source gate GEMM (f16 weights, TDM double-buffered) --------
__global__ __launch_bounds__(256) void k_gemm3(
    const float* __restrict__ X0, const _Float16* __restrict__ W0,
    const float* __restrict__ X1, const _Float16* __restrict__ W1,
    const float* __restrict__ X2, const _Float16* __restrict__ W2,
    const float* __restrict__ b0, const float* __restrict__ b1,
    const float* __restrict__ b2, float* __restrict__ Out,
    int M, int N, int K)
{
  __shared__ __align__(16) _Float16 Xs[128 * LDP];
  __shared__ __align__(16) _Float16 Ws[2][64 * LDP];
  const int tid  = threadIdx.x;
  const int lane = tid & 31;
  const int wv   = tid >> 5;
  const int mb   = blockIdx.y * 128;
  const int nb   = blockIdx.x * 64;
  const unsigned ws_off0 = (unsigned)(size_t)(&Ws[0][0]);
  const unsigned ws_off1 = (unsigned)(size_t)(&Ws[1][0]);
  const int KSTEPS = K >> 5;           // 16
  const int NSTEP  = 3 * KSTEPS;       // 48 flat steps
  v8f acc[4];
  { v8f z = {};
#pragma unroll
    for (int t = 0; t < 4; ++t) acc[t] = z; }

  if (wv == 0)
    tdm_load_tile_f16(ws_off0, W0 + (size_t)nb * K, (unsigned)K,
                      (unsigned)(N - nb), 64u);
  int buf = 0;
  for (int st = 0; st < NSTEP; ++st) {
    int seg = st / KSTEPS, kk = (st % KSTEPS) << 5;
    if (wv == 0) {
      if (st + 1 < NSTEP) {
        int sg2 = (st + 1) / KSTEPS, kk2 = ((st + 1) % KSTEPS) << 5;
        const _Float16* Wn = (sg2 == 0) ? W0 : (sg2 == 1) ? W1 : W2;
        tdm_load_tile_f16(buf ? ws_off0 : ws_off1,
                          Wn + (size_t)nb * K + kk2, (unsigned)K,
                          (unsigned)(N - nb), 64u);
        __builtin_amdgcn_s_wait_tensorcnt(1);
      } else {
        __builtin_amdgcn_s_wait_tensorcnt(0);
      }
    }
    const float* Xc = (seg == 0) ? X0 : (seg == 1) ? X1 : X2;
    load_tile<128>(Xc, K, mb, kk, Xs, tid);
    __syncthreads();
    v16h af = frag_a(&Xs[wv * 16 * LDP], lane);
    const _Float16* wsb = &Ws[buf][0];
#pragma unroll
    for (int nt = 0; nt < 4; ++nt) {
      v16h bf = frag_b(wsb + nt * 16 * LDP, lane);
      acc[nt] = wmma_f16(af, bf, acc[nt]);
    }
    __syncthreads();
    buf ^= 1;
  }

  const int nloc = lane & 15;
  const int moff = (lane < 16) ? 0 : 8;
#pragma unroll
  for (int nt = 0; nt < 4; ++nt) {
    int gn = nb + nt * 16 + nloc;
    if (gn < N) {
      float bv = b0[gn] + b1[gn] + b2[gn];
#pragma unroll
      for (int j = 0; j < 8; ++j) {
        int gm = mb + wv * 16 + j + moff;
        Out[(size_t)gm * N + gn] = acc[nt][j] + bv;
      }
    }
  }
}

// ---------------- batched attention scores (all-TDM staging, double-buffered) --------
// scores[b,a] = sum_o tanh( (att[b]@Wa^T)[a,o] + ba[o] + ah[b,a] ) * Wd[o] + bd
__global__ __launch_bounds__(256) void k_att_scores(
    const _Float16* __restrict__ att16,  // B,A,R (f16)
    const _Float16* __restrict__ Wa16,   // A,R   (f16)
    const float* __restrict__ ba,        // A
    const float* __restrict__ ahv,       // B,A   (h@Wh^T + bh)
    const float* __restrict__ Wd,        // A
    const float* __restrict__ bd,        // 1
    float* __restrict__ scores)          // B,A
{
  __shared__ __align__(16) _Float16 Xs[2][64 * LDP];
  __shared__ __align__(16) _Float16 Ws[2][224 * LDP];
  __shared__ float ahs[64];
  __shared__ float wds[208];
  __shared__ float bas[208];
  __shared__ float sc[64];
  const int tid = threadIdx.x, lane = tid & 31, wv = tid >> 5;
  const int b   = blockIdx.y;
  const int mb  = blockIdx.x * 64;
  const unsigned xs_off0 = (unsigned)(size_t)(&Xs[0][0]);
  const unsigned xs_off1 = (unsigned)(size_t)(&Xs[1][0]);
  const unsigned ws_off0 = (unsigned)(size_t)(&Ws[0][0]);
  const unsigned ws_off1 = (unsigned)(size_t)(&Ws[1][0]);

  for (int i = tid; i < 208; i += 256) {
    wds[i] = (i < A_) ? Wd[i] : 0.f;
    bas[i] = (i < A_) ? ba[i] : 0.f;
  }
  if (tid < 64) {
    int m = mb + tid;
    ahs[tid] = (m < A_) ? ahv[(size_t)b * A_ + m] : 0.f;
    sc[tid]  = 0.f;
  }

  v8f acc[7];
  { v8f z = {};
#pragma unroll
    for (int t = 0; t < 7; ++t) acc[t] = z; }

  const _Float16* Xb = att16 + (size_t)b * A_ * R_ + (size_t)mb * R_;

  if (wv == 0) {   // prologue pair into buffer 0
    tdm_load_tile_f16(xs_off0, Xb, (unsigned)R_, (unsigned)(A_ - mb), 64u);
    tdm_load_tile_f16(ws_off0, Wa16, (unsigned)R_, (unsigned)A_, 224u);
  }
  int buf = 0;
  for (int kk = 0; kk < R_; kk += 32) {
    if (wv == 0) {
      if (kk + 32 < R_) {
        tdm_load_tile_f16(buf ? xs_off0 : xs_off1, Xb + kk + 32,
                          (unsigned)R_, (unsigned)(A_ - mb), 64u);
        tdm_load_tile_f16(buf ? ws_off0 : ws_off1, Wa16 + kk + 32,
                          (unsigned)R_, (unsigned)A_, 224u);
        __builtin_amdgcn_s_wait_tensorcnt(2);   // current pair landed
      } else {
        __builtin_amdgcn_s_wait_tensorcnt(0);
      }
    }
    __syncthreads();
    const _Float16* xsb = &Xs[buf][0];
    const _Float16* wsb = &Ws[buf][0];
#pragma unroll
    for (int t = 0; t < 7; ++t) {          // wave-uniform tile assignment
      int g = wv + (t << 3);
      if (g < 52) {
        int mt = g & 3, nt = g >> 2;
        v16h af = frag_a(xsb + mt * 16 * LDP, lane);
        v16h bf = frag_b(wsb + nt * 16 * LDP, lane);
        acc[t] = wmma_f16(af, bf, acc[t]);
      }
    }
    __syncthreads();
    buf ^= 1;
  }

  const int nloc = lane & 15;
  const int moff = (lane < 16) ? 0 : 8;
#pragma unroll
  for (int t = 0; t < 7; ++t) {
    int g = wv + (t << 3);
    if (g >= 52) continue;
    int mt = g & 3, nt = g >> 2;
    int n = nt * 16 + nloc;
    float bav = bas[n], wdv = wds[n];
#pragma unroll
    for (int j = 0; j < 8; ++j) {
      int ml = mt * 16 + j + moff;
      int gm = mb + ml;
      float v = 0.f;
      if (n < A_ && gm < A_)
        v = tanhf(acc[t][j] + bav + ahs[ml]) * wdv;
#pragma unroll
      for (int s2 = 1; s2 < 16; s2 <<= 1) v += __shfl_xor(v, s2, 32);
      if (nloc == 0) atomicAdd(&sc[ml], v);   // ds_add_f32
    }
  }
  __syncthreads();
  if (tid < 64) {
    int m = mb + tid;
    if (m < A_) scores[(size_t)b * A_ + m] = sc[tid] + bd[0];
  }
}

// ---------------- softmax over A + weighted sum over f16 att (optionally + h) --------
__global__ __launch_bounds__(256) void k_softmax_wsum(
    const float* __restrict__ scores, const _Float16* __restrict__ att16,
    const float* __restrict__ addh, float* __restrict__ out)  // out: B,R
{
  __shared__ float wsm[A_];
  __shared__ float red[8];
  const int b = blockIdx.x, tid = threadIdx.x;
  float s = (tid < A_) ? scores[(size_t)b * A_ + tid] : -INFINITY;
  float mx = blk_reduce(s, true, red);
  float e  = (tid < A_) ? expf(s - mx) : 0.f;
  float sm = blk_reduce(e, false, red);
  if (tid < A_) wsm[tid] = e / sm;
  __syncthreads();
#pragma unroll
  for (int rr = 0; rr < 2; ++rr) {
    int r = tid + rr * 256;
    float acc = 0.f;
    for (int a = 0; a < A_; ++a)
      acc += (float)att16[((size_t)b * A_ + a) * R_ + r] * wsm[a];
    out[(size_t)b * R_ + r] = acc + (addh ? addh[(size_t)b * R_ + r] : 0.f);
  }
}

// ---------------- LSTM elementwise ----------------
__global__ __launch_bounds__(256) void k_lstm(
    const float* __restrict__ sums, const float* __restrict__ prev_c,
    float* __restrict__ next_c, float* __restrict__ next_h)
{
  int idx = blockIdx.x * 256 + threadIdx.x;   // b*R + r
  int b = idx / R_, r = idx - b * R_;
  const float* row = sums + (size_t)b * H4_;
  float ig = 1.f / (1.f + expf(-row[r]));
  float fg = 1.f / (1.f + expf(-row[R_ + r]));
  float og = 1.f / (1.f + expf(-row[2 * R_ + r]));
  float it = tanhf(row[3 * R_ + r]);
  float nc = fg * prev_c[idx] + ig * it;
  next_c[idx] = nc;
  next_h[idx] = og * tanhf(nc);
}

// ---------------- log-softmax over V ----------------
__global__ __launch_bounds__(256) void k_logsoftmax(
    const float* __restrict__ logits, float* __restrict__ out)
{
  __shared__ float red[8];
  const int b = blockIdx.x, tid = threadIdx.x;
  const float* row = logits + (size_t)b * V_;
  float mx = -INFINITY;
  for (int i = tid; i < V_; i += 256) mx = fmaxf(mx, row[i]);
  mx = blk_reduce(mx, true, red);
  float sm = 0.f;
  for (int i = tid; i < V_; i += 256) sm += expf(row[i] - mx);
  sm = blk_reduce(sm, false, red);
  float lse = mx + logf(sm);
  float* orow = out + (size_t)b * V_;
  for (int i = tid; i < V_; i += 256) orow[i] = row[i] - lse;
}

// ---------------- host launch ----------------
extern "C" void kernel_launch(void* const* d_in, const int* in_sizes, int n_in,
                              void* d_out, int out_size, void* d_ws, size_t ws_size,
                              hipStream_t stream)
{
  (void)in_sizes; (void)n_in; (void)out_size; (void)ws_size;
  const float* x      = (const float*)d_in[0];
  const float* att    = (const float*)d_in[1];
  const float* prev_c = (const float*)d_in[2];
  const float* prev_h = (const float*)d_in[3];
  const float* W_a2a  = (const float*)d_in[4];
  const float* b_a2a  = (const float*)d_in[5];
  const float* W_h2a  = (const float*)d_in[6];
  const float* b_h2a  = (const float*)d_in[7];
  const float* W_d2d  = (const float*)d_in[8];
  const float* b_d2d  = (const float*)d_in[9];
  const float* W_i2h  = (const float*)d_in[10];
  const float* b_i2h  = (const float*)d_in[11];
  const float* W_a2h  = (const float*)d_in[12];
  const float* b_a2h  = (const float*)d_in[13];
  const float* W_h2h  = (const float*)d_in[14];
  const float* b_h2h  = (const float*)d_in[15];
  const float* W_a2a1 = (const float*)d_in[16];
  const float* b_a2a1 = (const float*)d_in[17];
  const float* W_h2a1 = (const float*)d_in[18];
  const float* b_h2a1 = (const float*)d_in[19];
  const float* W_d2d1 = (const float*)d_in[20];
  const float* b_d2d1 = (const float*)d_in[21];
  const float* W_proj = (const float*)d_in[22];
  const float* b_proj = (const float*)d_in[23];

  // f32 scratch
  float* ws     = (float*)d_ws;
  float* sc     = ws;                 // B*A scores
  float* ah     = sc + B_ * A_;       // B*A  h-projection
  float* ar1    = ah + B_ * A_;       // B*R  attention result 1
  float* sums   = ar1 + B_ * R_;      // B*4R gate pre-activations
  float* nh     = sums + B_ * H4_;    // B*R  next_h
  float* logits = nh + B_ * R_;       // B*V
  // f16 scratch (TDM-fed operands)
  _Float16* h16      = (_Float16*)(logits + (size_t)B_ * V_);
  _Float16* att16    = h16;                                    // B*A*R
  _Float16* wa2a16   = att16  + (size_t)B_ * A_ * R_;          // A*R
  _Float16* wh2a16   = wa2a16 + (size_t)A_ * R_;               // A*R
  _Float16* wa2a116  = wh2a16 + (size_t)A_ * R_;               // A*R
  _Float16* wh2a116  = wa2a116 + (size_t)A_ * R_;              // A*R
  _Float16* wi2h16   = wh2a116 + (size_t)A_ * R_;              // 4R*R
  _Float16* wh2h16   = wi2h16 + (size_t)H4_ * R_;              // 4R*R
  _Float16* wa2h16   = wh2h16 + (size_t)H4_ * R_;              // 4R*R
  _Float16* wproj16  = wa2h16 + (size_t)H4_ * R_;              // V*R

  float* out_c  = (float*)d_out;          // next_c
  float* out_h  = out_c + B_ * R_;        // top_h
  float* out_ls = out_h + B_ * R_;        // log-softmax

  dim3 blk(256);

  // ---- one-time f32 -> f16 conversions (TDM/B-side operands) ----
  k_f32_to_f16<<<dim3(B_ * A_ * R_ / 2048), blk, 0, stream>>>(att, att16, B_ * A_ * R_);
  k_f32_to_f16<<<dim3(A_ * R_ / 2048), blk, 0, stream>>>(W_a2a,  wa2a16,  A_ * R_);
  k_f32_to_f16<<<dim3(A_ * R_ / 2048), blk, 0, stream>>>(W_h2a,  wh2a16,  A_ * R_);
  k_f32_to_f16<<<dim3(A_ * R_ / 2048), blk, 0, stream>>>(W_a2a1, wa2a116, A_ * R_);
  k_f32_to_f16<<<dim3(A_ * R_ / 2048), blk, 0, stream>>>(W_h2a1, wh2a116, A_ * R_);
  k_f32_to_f16<<<dim3(H4_ * R_ / 2048), blk, 0, stream>>>(W_i2h, wi2h16, H4_ * R_);
  k_f32_to_f16<<<dim3(H4_ * R_ / 2048), blk, 0, stream>>>(W_h2h, wh2h16, H4_ * R_);
  k_f32_to_f16<<<dim3(H4_ * R_ / 2048), blk, 0, stream>>>(W_a2h, wa2h16, H4_ * R_);
  k_f32_to_f16<<<dim3(V_ * R_ / 2048), blk, 0, stream>>>(W_proj, wproj16, V_ * R_);

  // ---- attention 1 (uses prev_h) ----
  k_gemm<<<dim3((A_ + 63) / 64, B_ / 128), blk, 0, stream>>>(
      prev_h, wh2a16, b_h2a, ah, B_, A_, R_);
  k_att_scores<<<dim3((A_ + 63) / 64, B_), blk, 0, stream>>>(
      att16, wa2a16, b_a2a, ah, W_d2d, b_d2d, sc);
  k_softmax_wsum<<<dim3(B_), blk, 0, stream>>>(sc, att16, nullptr, ar1);

  // ---- LSTM gates + elementwise ----
  k_gemm3<<<dim3(H4_ / 64, B_ / 128), blk, 0, stream>>>(
      x, wi2h16, prev_h, wh2h16, ar1, wa2h16,
      b_i2h, b_h2h, b_a2h, sums, B_, H4_, R_);
  k_lstm<<<dim3(B_ * R_ / 256), blk, 0, stream>>>(sums, prev_c, out_c, nh);

  // ---- attention 2 (uses next_h), top_h = wsum + next_h ----
  k_gemm<<<dim3((A_ + 63) / 64, B_ / 128), blk, 0, stream>>>(
      nh, wh2a116, b_h2a1, ah, B_, A_, R_);
  k_att_scores<<<dim3((A_ + 63) / 64, B_), blk, 0, stream>>>(
      att16, wa2a116, b_a2a1, ah, W_d2d1, b_d2d1, sc);
  k_softmax_wsum<<<dim3(B_), blk, 0, stream>>>(sc, att16, nh, out_h);

  // ---- projection + log-softmax ----
  k_gemm<<<dim3((V_ + 63) / 64, B_ / 128), blk, 0, stream>>>(
      out_h, wproj16, b_proj, logits, B_, V_, R_);
  k_logsoftmax<<<dim3(B_), blk, 0, stream>>>(logits, out_ls);
}
